// RNN_21723944583499
// MI455X (gfx1250) — matrix operand order repaired and checked by reading, not compiled, not run on hardware
//
#include <hip/hip_runtime.h>
#include <hip/hip_bf16.h>
#include <math.h>

// ---------------------------------------------------------------------------
// Problem constants (from the reference)
// ---------------------------------------------------------------------------
#define T_STEPS 512
#define BATCH   128
#define DIM_IN  512
#define HID     1024
#define OUTD    512

typedef __bf16 bf16;
typedef __attribute__((ext_vector_type(16))) __bf16 v16bf;
typedef __attribute__((ext_vector_type(8)))  __bf16 v8bf;
typedef __attribute__((ext_vector_type(8)))  float  v8f;
typedef __attribute__((ext_vector_type(4)))  float  v4f;

// ---------------------------------------------------------------------------
// WMMA fragment loaders (wave32, 16x16x32 bf16 shape)
//
// A fragment (16x32, bf16): lane m = lane&15.
//   lanes 0-15 : elems 0-7 = K[k0..k0+7],    elems 8-15 = K[k0+16..k0+23]
//   lanes 16-31: elems 0-7 = K[k0+8..k0+15], elems 8-15 = K[k0+24..k0+31]
// -> two contiguous 8-element chunks at (k0 + half*8) and (k0 + 16 + half*8).
// Source is f32; convert to bf16 in-register (v_cvt_pk_bf16_f32).
// ---------------------------------------------------------------------------
__device__ __forceinline__ v16bf load_fragA_f32(const float* __restrict__ A,
                                                int lda, int m0, int k0, int lane) {
    const int half = lane >> 4;
    const int m    = m0 + (lane & 15);
    const float* p = A + (size_t)m * lda + (size_t)(k0 + half * 8);
    v4f a0 = *(const v4f*)(p);
    v4f a1 = *(const v4f*)(p + 4);
    v4f a2 = *(const v4f*)(p + 16);
    v4f a3 = *(const v4f*)(p + 20);
    v16bf r;
#pragma unroll
    for (int i = 0; i < 4; ++i) {
        r[i]      = (__bf16)a0[i];
        r[4 + i]  = (__bf16)a1[i];
        r[8 + i]  = (__bf16)a2[i];
        r[12 + i] = (__bf16)a3[i];
    }
    return r;
}

// B fragment (32x16, bf16) from weight matrix W stored [N,K] row-major:
//   B_wmma[k,n] = W[n,k];  lane n = lane&15.
//   lanes 0-15: elems 0-7 = K[k0..k0+7], elems 8-15 = K[k0+16..k0+23];
//   lanes 16-31: elems 0-7 = K[k0+8..k0+15], elems 8-15 = K[k0+24..k0+31].
// -> two contiguous 16-byte runs per lane straight from global memory.
__device__ __forceinline__ v16bf load_fragB_bf16(const bf16* __restrict__ W,
                                                 int ldw, int n0, int k0, int lane) {
    const int half = lane >> 4;
    const int n    = n0 + (lane & 15);
    const bf16* p  = W + (size_t)n * ldw + (size_t)(k0 + half * 8);
    v8bf b0 = *(const v8bf*)(p);
    v8bf b1 = *(const v8bf*)(p + 16);
    v16bf r;
#pragma unroll
    for (int i = 0; i < 8; ++i) { r[i] = b0[i]; r[8 + i] = b1[i]; }
    return r;
}

// Fast activations built on HW transcendentals (v_exp_f32 / v_rcp_f32).
__device__ __forceinline__ float fast_sigmoid(float x) {
    return __builtin_amdgcn_rcpf(1.0f + __expf(-x));
}
__device__ __forceinline__ float fast_tanh(float x) {
    // tanh(x) = 1 - 2/(exp(2x)+1)
    return 1.0f - 2.0f * __builtin_amdgcn_rcpf(__expf(2.0f * x) + 1.0f);
}

// ---------------------------------------------------------------------------
// Tiled WMMA GEMM:  C[M,N] = epilogue( A[M,K](f32) @ W[N,K]^T(bf16) )
//
// Template blocking: each wave owns IM x IN WMMA tiles (IM*16 x IN*16 output).
// Block = 128 threads = 4 waves laid out along N. Block tile = IM*16 x IN*64.
// MODE (compile-time):
//   0: C = acc + bias[n]
//   1: C = sigmoid(acc + bias[n])
//   2: C = tanh(acc + C_old)   (in-place RNN step; xproj preloaded in C)
// ---------------------------------------------------------------------------
template <int IM, int IN, int MODE>
__global__ __launch_bounds__(128) void wmma_gemm(
    const float* __restrict__ A, const bf16* __restrict__ W,
    const float* __restrict__ bias, float* __restrict__ C,
    int M, int N, int K)
{
    const int lane = threadIdx.x & 31;
    const int wave = threadIdx.x >> 5;
    const int m0   = blockIdx.x * (IM * 16);
    const int n0   = blockIdx.y * (4 * IN * 16) + wave * (IN * 16);

    v8f acc[IM][IN];
#pragma unroll
    for (int im = 0; im < IM; ++im)
#pragma unroll
        for (int in = 0; in < IN; ++in) acc[im][in] = (v8f){};

#pragma unroll 2
    for (int k0 = 0; k0 < K; k0 += 32) {
        // Prefetch the next K-chunk of the weight rows (global_prefetch_b8);
        // weights are the reused operand and live in L2.
        if (k0 + 32 < K) {
#pragma unroll
            for (int in = 0; in < IN; ++in)
                __builtin_prefetch(
                    W + (size_t)(n0 + in * 16 + (lane & 15)) * K + (k0 + 32), 0, 3);
        }

        v16bf afrag[IM];
        v16bf bfrag[IN];
#pragma unroll
        for (int im = 0; im < IM; ++im)
            afrag[im] = load_fragA_f32(A, K, m0 + im * 16, k0, lane);
#pragma unroll
        for (int in = 0; in < IN; ++in)
            bfrag[in] = load_fragB_bf16(W, K, n0 + in * 16, k0, lane);

#pragma unroll
        for (int im = 0; im < IM; ++im)
#pragma unroll
            for (int in = 0; in < IN; ++in)
                acc[im][in] = __builtin_amdgcn_wmma_f32_16x16x32_bf16(
                    false, afrag[im], false, bfrag[in],
                    (short)0, acc[im][in], false, false);
    }

    // Epilogue. C/D layout: VGPR r, lanes 0-15 -> (m = r, n = lane);
    // lanes 16-31 -> (m = 8 + r, n = lane - 16).
    const int half = lane >> 4;
    const int nn   = lane & 15;

#pragma unroll
    for (int im = 0; im < IM; ++im) {
#pragma unroll
        for (int in = 0; in < IN; ++in) {
            const int nc = n0 + in * 16 + nn;
#pragma unroll
            for (int r = 0; r < 8; ++r) {
                const int mc = m0 + im * 16 + half * 8 + r;
                const size_t idx = (size_t)mc * N + nc;
                float v = acc[im][in][r];
                if constexpr (MODE == 2) {
                    v = fast_tanh(v + C[idx]);       // C holds xproj_t
                } else if constexpr (MODE == 1) {
                    v = fast_sigmoid(v + bias[nc]);
                } else {
                    v += bias[nc];
                }
                C[idx] = v;
            }
        }
    }
}

// ---------------------------------------------------------------------------
// Small helpers
// ---------------------------------------------------------------------------
__global__ void cvt_f32_to_bf16_kernel(const float* __restrict__ src,
                                       bf16* __restrict__ dst, int n) {
    int i = (blockIdx.x * blockDim.x + threadIdx.x) * 4;
    if (i + 3 < n) {
        v4f s = *(const v4f*)(src + i);
        bf16 d[4];
#pragma unroll
        for (int j = 0; j < 4; ++j) d[j] = (bf16)s[j];
        *(uint64_t*)(dst + i) = *(const uint64_t*)d;
    } else {
        for (; i < n; ++i) dst[i] = (bf16)src[i];
    }
}

__global__ void add_vec_kernel(const float* __restrict__ a,
                               const float* __restrict__ b,
                               float* __restrict__ o, int n) {
    int i = blockIdx.x * blockDim.x + threadIdx.x;
    if (i < n) o[i] = a[i] + b[i];
}

// ---------------------------------------------------------------------------
// kernel_launch
// ---------------------------------------------------------------------------
extern "C" void kernel_launch(void* const* d_in, const int* in_sizes, int n_in,
                              void* d_out, int out_size, void* d_ws, size_t ws_size,
                              hipStream_t stream) {
    const float* x    = (const float*)d_in[0]; // [T,B,DIM]
    const float* h0   = (const float*)d_in[1]; // [B,H]
    const float* Wx_w = (const float*)d_in[2]; // [H,DIM]
    const float* Wx_b = (const float*)d_in[3]; // [H]
    const float* Wh_w = (const float*)d_in[4]; // [H,H]
    const float* Wh_b = (const float*)d_in[5]; // [H]
    const float* Wo_w = (const float*)d_in[6]; // [OUT,H]
    const float* Wo_b = (const float*)d_in[7]; // [OUT]

    // Output layout: all_h [(T+1),B,H] | all_y [T,B,OUT] | last_logits [B,OUT]
    float* all_h = (float*)d_out;
    float* all_y = all_h + (size_t)(T_STEPS + 1) * BATCH * HID;
    float* last  = all_y + (size_t)T_STEPS * BATCH * OUTD;

    // Workspace: bf16 weight mirrors + combined bias (~4.2 MB)
    char* ws = (char*)d_ws;
    bf16*  Wx_bf  = (bf16*)ws;  ws += (size_t)HID * DIM_IN * sizeof(bf16);
    bf16*  Wh_bf  = (bf16*)ws;  ws += (size_t)HID * HID * sizeof(bf16);
    bf16*  Wo_bf  = (bf16*)ws;  ws += (size_t)OUTD * HID * sizeof(bf16);
    float* bias_c = (float*)ws; ws += (size_t)HID * sizeof(float);

    const int ct = 256;
    cvt_f32_to_bf16_kernel<<<(HID * DIM_IN / 4 + ct - 1) / ct, ct, 0, stream>>>(
        Wx_w, Wx_bf, HID * DIM_IN);
    cvt_f32_to_bf16_kernel<<<(HID * HID / 4 + ct - 1) / ct, ct, 0, stream>>>(
        Wh_w, Wh_bf, HID * HID);
    cvt_f32_to_bf16_kernel<<<(OUTD * HID / 4 + ct - 1) / ct, ct, 0, stream>>>(
        Wo_w, Wo_bf, OUTD * HID);
    add_vec_kernel<<<(HID + ct - 1) / ct, ct, 0, stream>>>(Wx_b, Wh_b, bias_c, HID);

    // all_h[0] = h0
    hipMemcpyAsync(all_h, h0, (size_t)BATCH * HID * sizeof(float),
                   hipMemcpyDeviceToDevice, stream);

    // xproj: Wx@x + (Wx_b+Wh_b) written directly into all_h[1..T]
    // (consumed in place by the step kernels). 2x4 tiles per wave.
    wmma_gemm<2, 4, 0><<<dim3((T_STEPS * BATCH) / 32, HID / 256), 128, 0, stream>>>(
        x, Wx_bf, bias_c, all_h + (size_t)BATCH * HID, T_STEPS * BATCH, HID, DIM_IN);

    // Recurrence: h_t = tanh(h_{t-1} @ Wh^T + xproj_t), in place over all_h[t+1].
    // 1x2 tiles per wave -> 256 waves per step for latency coverage.
    for (int t = 0; t < T_STEPS; ++t) {
        wmma_gemm<1, 2, 2><<<dim3(BATCH / 16, HID / 128), 128, 0, stream>>>(
            all_h + (size_t)t * BATCH * HID, Wh_bf, nullptr,
            all_h + (size_t)(t + 1) * BATCH * HID, BATCH, HID, HID);
    }

    // Decode: all_y = sigmoid(hs @ Wo^T + Wo_b), batched over all timesteps.
    wmma_gemm<2, 4, 1><<<dim3((T_STEPS * BATCH) / 32, OUTD / 256), 128, 0, stream>>>(
        all_h + (size_t)BATCH * HID, Wo_bf, Wo_b, all_y, T_STEPS * BATCH, OUTD, HID);

    // last_logits = h_T @ Wo^T + Wo_b (no activation)
    wmma_gemm<2, 4, 0><<<dim3(BATCH / 32, OUTD / 256), 128, 0, stream>>>(
        all_h + (size_t)T_STEPS * BATCH * HID, Wo_bf, Wo_b, last, BATCH, OUTD, HID);
}
